// MultiHeadNeuralMemoryMLP_26989574488238
// MI455X (gfx1250) — matrix operand order repaired and checked by reading, not compiled, not run on hardware
//
#include <hip/hip_runtime.h>

#define NH_C 16
#define L_TOK 1024
#define DIM 1024

typedef __attribute__((ext_vector_type(16))) __bf16 v16bf;
typedef __attribute__((ext_vector_type(8)))  __bf16 bh8;
typedef __attribute__((ext_vector_type(8)))  float  v8f;

__device__ __forceinline__ float sigf(float x) { return 1.f / (1.f + __expf(-x)); }

// Build a 16-half WMMA operand from two 16-byte LDS vectors.
__device__ __forceinline__ v16bf ld_frag2(const __bf16* plo, const __bf16* phi) {
  bh8 lo = *(const bh8*)plo;
  bh8 hi = *(const bh8*)phi;
  return __builtin_shufflevector(lo, hi, 0, 1, 2, 3, 4, 5, 6, 7,
                                 8, 9, 10, 11, 12, 13, 14, 15);
}

// ============================ elementwise kernels ============================

__global__ void k_f2b(const float* __restrict__ s, __bf16* __restrict__ d, long n) {
  long i = (long)blockIdx.x * blockDim.x + threadIdx.x;
  if (i < n) d[i] = (__bf16)s[i];
}

__global__ void k_silu_fwd(const float* __restrict__ z, __bf16* __restrict__ xb,
                           float* __restrict__ ds, long n) {
  long i = (long)blockIdx.x * blockDim.x + threadIdx.x;
  if (i >= n) return;
  float v = z[i], s = sigf(v), sil = v * s;
  xb[i] = (__bf16)sil;
  ds[i] = sil + s * (1.f - sil);
}

__global__ void k_silu(const float* __restrict__ z, __bf16* __restrict__ xb, long n) {
  long i = (long)blockIdx.x * blockDim.x + threadIdx.x;
  if (i >= n) return;
  float v = z[i];
  xb[i] = (__bf16)(v * sigf(v));
}

// grad_Z2 = Z2 - V_head ; v2 = lr * grad_Z2   (n = Z*L*64)
__global__ void k_gz2(const float* __restrict__ Z2, const float* __restrict__ Vf,
                      const float* __restrict__ lr, __bf16* __restrict__ gz2b,
                      __bf16* __restrict__ v2b, long n) {
  long i = (long)blockIdx.x * blockDim.x + threadIdx.x;
  if (i >= n) return;
  int  d = (int)(i & 63);
  long t = i >> 6;
  int  l = (int)(t & (L_TOK - 1));
  long z = t >> 10;
  int  b = (int)(z >> 4), h = (int)(z & 15);
  float v = Vf[(long)b * L_TOK * DIM + (long)l * DIM + h * 64 + d];
  float g = Z2[i] - v;
  gz2b[i] = (__bf16)g;
  v2b[i]  = (__bf16)(lr[z * L_TOK + l] * g);
}

// grad_Z1 = dsilu * grad_X2 ; v1 = lr * grad_Z1  (n = Z*L*128)
__global__ void k_gz1(const float* __restrict__ gX2, const float* __restrict__ ds,
                      const float* __restrict__ lr, __bf16* __restrict__ g1b,
                      __bf16* __restrict__ v1b, long n) {
  long i = (long)blockIdx.x * blockDim.x + threadIdx.x;
  if (i >= n) return;
  int  l = (int)((i >> 7) & (L_TOK - 1));
  long z = i >> 17;
  float g = ds[i] * gX2[i];
  g1b[i] = (__bf16)g;
  v1b[i] = (__bf16)(lr[z * L_TOK + l] * g);
}

// u = lr * exp(cumN - cum) * g  (feature size = 1<<logF)
__global__ void k_u(const __bf16* __restrict__ g, const float* __restrict__ lr,
                    const float* __restrict__ cum, __bf16* __restrict__ u, long n, int logF) {
  long i = (long)blockIdx.x * blockDim.x + threadIdx.x;
  if (i >= n) return;
  int  l = (int)((i >> logF) & (L_TOK - 1));
  long z = i >> (logF + 10);
  float cn = cum[z * L_TOK + (L_TOK - 1)];
  u[i] = (__bf16)(lr[z * L_TOK + l] * __expf(cn - cum[z * L_TOK + l]) * (float)g[i]);
}

// W_next = -raw + W * exp(cumN)   (8192 elems per head)
__global__ void k_wnext(const float* __restrict__ raw, const float* __restrict__ W,
                        const float* __restrict__ cum, float* __restrict__ out, long n) {
  long i = (long)blockIdx.x * blockDim.x + threadIdx.x;
  if (i >= n) return;
  long z = i >> 13;
  out[i] = -raw[i] + W[i] * __expf(cum[z * L_TOK + (L_TOK - 1)]);
}

// [Z,L,64] -> [B,L,D] head-concat, bf16
__global__ void k_z2cat(const float* __restrict__ Z2_, __bf16* __restrict__ out, long n) {
  long i = (long)blockIdx.x * blockDim.x + threadIdx.x;
  if (i >= n) return;
  int  d = (int)(i & 63);
  long t = i >> 6;
  int  l = (int)(t & (L_TOK - 1));
  long z = t >> 10;
  int  b = (int)(z >> 4), h = (int)(z & 15);
  out[(long)b * L_TOK * DIM + (long)l * DIM + h * 64 + d] = (__bf16)Z2_[i];
}

// per-(b,n): lr[l] and inclusive cumsum of log(1 - wd*sigmoid(...))
__global__ void k_lrwd(const float* __restrict__ x, const float* __restrict__ lblr,
                       const float* __restrict__ flw, const float* __restrict__ flb,
                       const float* __restrict__ lbwd, const float* __restrict__ fww,
                       const float* __restrict__ fwb, float* __restrict__ lr,
                       float* __restrict__ cum) {
  __shared__ float slog[L_TOK];
  int z = blockIdx.x, b = z / NH_C, n = z % NH_C;
  const float* xb = x + (long)b * L_TOK * DIM;
  const float* wl = flw + (long)n * DIM;
  const float* ww = fww + (long)n * DIM;
  float base_lr = __expf(lblr[n]);
  float base_wd = __expf(lbwd[n]);
  float bl = flb[n], bw = fwb[n];
  for (int l = threadIdx.x; l < L_TOK; l += blockDim.x) {
    const float* xr = xb + (long)l * DIM;
    float a = 0.f, c = 0.f;
    for (int d = 0; d < DIM; d++) { float xv = xr[d]; a += xv * wl[d]; c += xv * ww[d]; }
    lr[(long)z * L_TOK + l] = base_lr * sigf(a + bl);
    slog[l] = log1pf(-base_wd * sigf(c + bw));
  }
  __syncthreads();
  if (threadIdx.x == 0) {
    float acc = 0.f;
    for (int l = 0; l < L_TOK; l++) { acc += slog[l]; cum[(long)z * L_TOK + l] = acc; }
  }
}

// ============================ batched WMMA GEMM ============================
// C[m,n] = sum_k A[m,k]*B[k,n] (+bias[n]); transA: A stored [K,M]; transB: B stored [N,K].
// B tile kept N-major in LDS (sBt[n][k]) so WMMA B-fragments are contiguous b128 loads.
// Batch z: offsets b*BB + n*BN (b=z/NH, n=z%NH) to support [B,L,D] head views.
__global__ __launch_bounds__(128) void k_gemm(
    const __bf16* __restrict__ A, const __bf16* __restrict__ Bm,
    float* __restrict__ C, const float* __restrict__ bias,
    int M, int N, int K, int lda, int ldb, int ldc,
    long aBB, long aBN, long bBB, long bBN, long cBB, long cBN,
    int transA, int transB) {
  (void)M; (void)N;
  __shared__ __align__(16) __bf16 sA[64][72];   // [m][k]  pitch 144B (16B mult)
  __shared__ __align__(16) __bf16 sBt[64][72];  // [n][k]  pitch 144B
  const int z = blockIdx.z, b = z / NH_C, n = z % NH_C;
  const long aOff = (long)b * aBB + (long)n * aBN;
  const long bOff = (long)b * bBB + (long)n * bBN;
  const long cOff = (long)b * cBB + (long)n * cBN;
  const int gm0 = blockIdx.y * 64, gn0 = blockIdx.x * 64;
  const int tid = threadIdx.x, lane = tid & 31, wave = tid >> 5;
  const int wm0 = wave * 16;
  const int mrow = lane & 15, ncol = lane & 15;
  const int kbA = (lane < 16) ? 0 : 8, kbB = (lane < 16) ? 0 : 16;
  v8f zero = {0.f, 0.f, 0.f, 0.f, 0.f, 0.f, 0.f, 0.f};
  v8f acc[4];
#pragma unroll
  for (int j = 0; j < 4; j++) acc[j] = zero;

  for (int k0 = 0; k0 < K; k0 += 64) {
    // ---- stage A (64 x 64 halves) ----
    if (!transA) {
#pragma unroll 4
      for (int idx = tid; idx < 64 * 8; idx += 128) {
        int m = idx >> 3, c = idx & 7;
        *(bh8*)&sA[m][c * 8] =
            *(const bh8*)&A[aOff + (long)(gm0 + m) * lda + k0 + c * 8];
      }
    } else {  // A stored [K,M]: read contiguous m, scatter into LDS
#pragma unroll 8
      for (int idx = tid; idx < 64 * 64; idx += 128) {
        int k = idx >> 6, m = idx & 63;
        sA[m][k] = A[aOff + (long)(k0 + k) * lda + (gm0 + m)];
      }
    }
    // ---- stage B transposed into sBt[n][k] ----
    if (transB) {  // B stored [N,K]: rows are contiguous in k
#pragma unroll 4
      for (int idx = tid; idx < 64 * 8; idx += 128) {
        int nn = idx >> 3, c = idx & 7;
        *(bh8*)&sBt[nn][c * 8] =
            *(const bh8*)&Bm[bOff + (long)(gn0 + nn) * ldb + k0 + c * 8];
      }
    } else {  // B stored [K,N]: read contiguous n, scatter
#pragma unroll 8
      for (int idx = tid; idx < 64 * 64; idx += 128) {
        int k = idx >> 6, nn = idx & 63;
        sBt[nn][k] = Bm[bOff + (long)(k0 + k) * ldb + (gn0 + nn)];
      }
    }
    if (k0 + 64 < K) {  // prime L2 for next k-tile -> global_prefetch_b8
      long o = transA ? (long)(k0 + 64 + (tid & 31)) * lda + gm0
                      : (long)(gm0 + (tid & 63)) * lda + (k0 + 64);
      __builtin_prefetch(&A[aOff + o], 0, 1);
    }
    __syncthreads();
#pragma unroll
    for (int ks = 0; ks < 64; ks += 32) {
      const __bf16* arow = &sA[wm0 + mrow][ks];
      v16bf af = ld_frag2(arow + kbA, arow + kbA + 16);
#pragma unroll
      for (int j = 0; j < 4; j++) {
        const __bf16* brow = &sBt[j * 16 + ncol][ks];
        v16bf bfr = ld_frag2(brow + kbA, brow + kbA + 16);
        acc[j] = __builtin_amdgcn_wmma_f32_16x16x32_bf16(false, af, false, bfr,
                                                         (short)0, acc[j], false, false);
      }
    }
    __syncthreads();
  }
#pragma unroll
  for (int j = 0; j < 4; j++) {
    int nn = gn0 + j * 16 + ncol;
    float bv = bias ? bias[nn] : 0.f;
#pragma unroll
    for (int r = 0; r < 8; r++) {
      int mm = gm0 + wm0 + r + ((lane < 16) ? 0 : 8);
      C[cOff + (long)mm * ldc + nn] = acc[j][r] + bv;
    }
  }
}

// ================= fused decayed linear attention (flash-style) =================
// Out[m,h] = Base[m,h]*exp(cum[m]) - sum_{l<=m} (q[m].k[l]) * exp(cum[m]-cum[l]) * v[l,h]
template <int DK, int DV>
__global__ __launch_bounds__(128) void k_attn(
    const __bf16* __restrict__ Qp, int ldq, long qBB, long qBN,
    const __bf16* __restrict__ Kp, int ldk, long kBB, long kBN,
    const __bf16* __restrict__ Vp, const float* __restrict__ Base,
    const float* __restrict__ cum, float* __restrict__ Out) {
  __shared__ __align__(16) __bf16 sQ[64][DK + 8];  // [m][d] pitch 16B-mult
  __shared__ __align__(16) __bf16 sK[64][DK + 8];  // [l][d]
  __shared__ __align__(16) __bf16 sVt[DV][72];     // [h][l] transposed
  __shared__ __align__(16) __bf16 sP[64][72];      // [m][l]
  __shared__ float scm[64], scl[64];

  const int z = blockIdx.y, b = z / NH_C, n = z % NH_C;
  const int mt = blockIdx.x, m0 = mt * 64;
  const int tid = threadIdx.x, lane = tid & 31, wave = tid >> 5;
  const long qOff = (long)b * qBB + (long)n * qBN;
  const long kOff = (long)b * kBB + (long)n * kBN;
  const float*  cz = cum + (long)z * L_TOK;
  const __bf16* Vz = Vp + (long)z * L_TOK * DV;
  const float*  Bz = Base + (long)z * L_TOK * DV;
  float*        Oz = Out + (long)z * L_TOK * DV;

#pragma unroll 4
  for (int idx = tid; idx < 64 * (DK / 8); idx += 128) {
    int m = idx / (DK / 8), c = idx % (DK / 8);
    *(bh8*)&sQ[m][c * 8] =
        *(const bh8*)&Qp[qOff + (long)(m0 + m) * ldq + c * 8];
  }
  for (int i = tid; i < 64; i += 128) scm[i] = cz[m0 + i];
  __syncthreads();

  const int wm0 = wave * 16;
  const int mrow = lane & 15, ncol = lane & 15;
  const int kbA = (lane < 16) ? 0 : 8, kbB = (lane < 16) ? 0 : 16;
  v8f zero = {0.f, 0.f, 0.f, 0.f, 0.f, 0.f, 0.f, 0.f};
  v8f oacc[DV / 16];
#pragma unroll
  for (int j = 0; j < DV / 16; j++) oacc[j] = zero;

  for (int lt = 0; lt <= mt; lt++) {
    const int l0 = lt * 64;
#pragma unroll 4
    for (int idx = tid; idx < 64 * (DK / 8); idx += 128) {
      int l = idx / (DK / 8), c = idx % (DK / 8);
      *(bh8*)&sK[l][c * 8] =
          *(const bh8*)&Kp[kOff + (long)(l0 + l) * ldk + c * 8];
    }
    // V tile, transposed into sVt[h][l]: coalesced global b128, scalar LDS scatter
#pragma unroll 4
    for (int idx = tid; idx < 64 * (DV / 8); idx += 128) {
      int l = idx / (DV / 8), c = idx % (DV / 8);
      bh8 v = *(const bh8*)&Vz[(long)(l0 + l) * DV + c * 8];
#pragma unroll
      for (int i = 0; i < 8; i++) sVt[c * 8 + i][l] = v[i];
    }
    for (int i = tid; i < 64; i += 128) scl[i] = cz[l0 + i];
    __syncthreads();

    // S = q @ k^T  (wave: 16 rows x 64 cols); B-frag rows of sK are contiguous
    v8f sacc[4];
#pragma unroll
    for (int j = 0; j < 4; j++) sacc[j] = zero;
#pragma unroll
    for (int ks = 0; ks < DK; ks += 32) {
      const __bf16* arow = &sQ[wm0 + mrow][ks];
      v16bf af = ld_frag2(arow + kbA, arow + kbA + 16);
#pragma unroll
      for (int j = 0; j < 4; j++) {
        const __bf16* brow = &sK[j * 16 + ncol][ks];
        v16bf bfr = ld_frag2(brow + kbA, brow + kbA + 16);
        sacc[j] = __builtin_amdgcn_wmma_f32_16x16x32_bf16(false, af, false, bfr,
                                                          (short)0, sacc[j], false, false);
      }
    }
    // decay + causal mask, P -> LDS (bf16)
#pragma unroll
    for (int j = 0; j < 4; j++) {
#pragma unroll
      for (int r = 0; r < 8; r++) {
        int ml = wm0 + r + ((lane < 16) ? 0 : 8);
        int ll = j * 16 + ncol;
        float f = ((l0 + ll) <= (m0 + ml)) ? __expf(scm[ml] - scl[ll]) : 0.f;
        sP[ml][ll] = (__bf16)(sacc[j][r] * f);
      }
    }
    __syncthreads();
    // O += P @ V   (A-frags from sP rows, B-frags from sVt rows: all contiguous)
#pragma unroll
    for (int ks = 0; ks < 64; ks += 32) {
      const __bf16* arow = &sP[wm0 + mrow][ks];
      v16bf af = ld_frag2(arow + kbA, arow + kbA + 16);
#pragma unroll
      for (int j = 0; j < DV / 16; j++) {
        const __bf16* brow = &sVt[j * 16 + ncol][ks];
        v16bf bfr = ld_frag2(brow + kbA, brow + kbA + 16);
        oacc[j] = __builtin_amdgcn_wmma_f32_16x16x32_bf16(false, af, false, bfr,
                                                          (short)0, oacc[j], false, false);
      }
    }
    __syncthreads();
  }
#pragma unroll
  for (int j = 0; j < DV / 16; j++) {
#pragma unroll
    for (int r = 0; r < 8; r++) {
      int ml = wm0 + r + ((lane < 16) ? 0 : 8);
      int h = j * 16 + ncol;
      long o = (long)(m0 + ml) * DV + h;
      Oz[o] = Bz[o] * __expf(scm[ml]) - oacc[j][r];
    }
  }
}

// ============================ host orchestration ============================

static inline void gemm(hipStream_t s, const __bf16* A, const __bf16* B, float* C,
                        const float* bias, int M, int N, int K, int lda, int ldb, int ldc,
                        long aBB, long aBN, long bBB, long bBN, long cBB, long cBN,
                        int tA, int tB, int batches) {
  dim3 g(N / 64, M / 64, batches);
  k_gemm<<<g, 128, 0, s>>>(A, B, C, bias, M, N, K, lda, ldb, ldc,
                           aBB, aBN, bBB, bBN, cBB, cBN, tA, tB);
}

static inline dim3 ew(long n) { return dim3((unsigned)((n + 255) / 256)); }

extern "C" void kernel_launch(void* const* d_in, const int* in_sizes, int n_in,
                              void* d_out, int out_size, void* d_ws, size_t ws_size,
                              hipStream_t stream) {
  (void)in_sizes; (void)n_in; (void)out_size; (void)ws_size;
  const float* x    = (const float*)d_in[0];
  const float* W1   = (const float*)d_in[1];
  const float* W2   = (const float*)d_in[2];
  const float* lblr = (const float*)d_in[3];
  const float* flw  = (const float*)d_in[4];
  const float* flb  = (const float*)d_in[5];
  const float* lbwd = (const float*)d_in[6];
  const float* fww  = (const float*)d_in[7];
  const float* fwb  = (const float*)d_in[8];
  const float* q_w  = (const float*)d_in[9];  const float* q_b = (const float*)d_in[10];
  const float* k_w  = (const float*)d_in[11]; const float* k_b = (const float*)d_in[12];
  const float* v_w  = (const float*)d_in[13]; const float* v_b = (const float*)d_in[14];
  const float* o_w  = (const float*)d_in[15]; const float* o_b = (const float*)d_in[16];
  float* out = (float*)d_out;
  float* W1n = out + 2097152;          // [2,16,128,64]
  float* W2n = W1n + 262144;           // [2,16,64,128]

  const long BLD = 2097152;            // B*L*D
  const long ZL128 = 4194304, ZL64 = 2097152;  // (B*NH)*L*{128,64}

  char* wsp = (char*)d_ws; size_t off = 0;
  auto alloc = [&](size_t bytes) -> void* {
    void* p = wsp + off; off += (bytes + 255) & ~(size_t)255; return p;
  };
  __bf16* xbf  = (__bf16*)alloc(BLD * 2);
  __bf16* qwbf = (__bf16*)alloc(1048576 * 2);
  __bf16* kwbf = (__bf16*)alloc(1048576 * 2);
  __bf16* vwbf = (__bf16*)alloc(1048576 * 2);
  __bf16* owbf = (__bf16*)alloc(1048576 * 2);
  __bf16* W1bf = (__bf16*)alloc(262144 * 2);
  __bf16* W2bf = (__bf16*)alloc(262144 * 2);
  float*  Qf   = (float*)alloc(BLD * 4);
  float*  Kf   = (float*)alloc(BLD * 4);
  float*  Vf   = (float*)alloc(BLD * 4);
  __bf16* Qbf  = (__bf16*)alloc(BLD * 2);
  __bf16* Kbf  = (__bf16*)alloc(BLD * 2);
  float*  lr   = (float*)alloc(32768 * 4);
  float*  cum  = (float*)alloc(32768 * 4);
  float*  Z1   = (float*)alloc(ZL128 * 4);
  __bf16* X2b  = (__bf16*)alloc(ZL128 * 2);
  float*  dsl  = (float*)alloc(ZL128 * 4);
  float*  Z2   = (float*)alloc(ZL64 * 4);
  __bf16* gz2b = (__bf16*)alloc(ZL64 * 2);
  __bf16* v2b  = (__bf16*)alloc(ZL64 * 2);
  float*  gX2  = (float*)alloc(ZL128 * 4);
  __bf16* g1b  = (__bf16*)alloc(ZL128 * 2);
  __bf16* v1b  = (__bf16*)alloc(ZL128 * 2);
  float*  Bas1 = (float*)alloc(ZL128 * 4);
  float*  Z1_  = (float*)alloc(ZL128 * 4);
  __bf16* X2_b = (__bf16*)alloc(ZL128 * 2);
  float*  Bas2 = (float*)alloc(ZL64 * 4);
  float*  Z2_  = (float*)alloc(ZL64 * 4);
  __bf16* u1b  = (__bf16*)alloc(ZL128 * 2);
  __bf16* u2b  = (__bf16*)alloc(ZL64 * 2);
  float*  W1r  = (float*)alloc(262144 * 4);
  float*  W2r  = (float*)alloc(262144 * 4);
  __bf16* z2c  = (__bf16*)alloc(BLD * 2);

  // bf16 conversions of inputs
  k_f2b<<<ew(BLD), 256, 0, stream>>>(x, xbf, BLD);
  k_f2b<<<ew(1048576), 256, 0, stream>>>(q_w, qwbf, 1048576);
  k_f2b<<<ew(1048576), 256, 0, stream>>>(k_w, kwbf, 1048576);
  k_f2b<<<ew(1048576), 256, 0, stream>>>(v_w, vwbf, 1048576);
  k_f2b<<<ew(1048576), 256, 0, stream>>>(o_w, owbf, 1048576);
  k_f2b<<<ew(262144), 256, 0, stream>>>(W1, W1bf, 262144);
  k_f2b<<<ew(262144), 256, 0, stream>>>(W2, W2bf, 262144);

  // lr / cumulative log-weight-decay
  k_lrwd<<<32, 256, 0, stream>>>(x, lblr, flw, flb, lbwd, fww, fwb, lr, cum);

  // Q,K,V projections  (M=2048, N=1024, K=1024)
  gemm(stream, xbf, qwbf, Qf, q_b, 2048, 1024, 1024, 1024, 1024, 1024, 0,0,0,0,0,0, 0, 1, 1);
  gemm(stream, xbf, kwbf, Kf, k_b, 2048, 1024, 1024, 1024, 1024, 1024, 0,0,0,0,0,0, 0, 1, 1);
  gemm(stream, xbf, vwbf, Vf, v_b, 2048, 1024, 1024, 1024, 1024, 1024, 0,0,0,0,0,0, 0, 1, 1);
  k_f2b<<<ew(BLD), 256, 0, stream>>>(Qf, Qbf, BLD);
  k_f2b<<<ew(BLD), 256, 0, stream>>>(Kf, Kbf, BLD);

  // Z1 = K_head @ W1^T  (batched over 32 heads; A = head view of [B,L,D])
  gemm(stream, Kbf, W1bf, Z1, nullptr, 1024, 128, 64, 1024, 64, 128,
       1048576, 64, 131072, 8192, 2097152, 131072, 0, 1, 32);
  k_silu_fwd<<<ew(ZL128), 256, 0, stream>>>(Z1, X2b, dsl, ZL128);

  // Z2 = X2 @ W2^T
  gemm(stream, X2b, W2bf, Z2, nullptr, 1024, 64, 128, 128, 128, 64,
       2097152, 131072, 131072, 8192, 1048576, 65536, 0, 1, 32);
  k_gz2<<<ew(ZL64), 256, 0, stream>>>(Z2, Vf, lr, gz2b, v2b, ZL64);

  // grad_X2 = grad_Z2 @ W2
  gemm(stream, gz2b, W2bf, gX2, nullptr, 1024, 128, 64, 64, 128, 128,
       1048576, 65536, 131072, 8192, 2097152, 131072, 0, 0, 32);
  k_gz1<<<ew(ZL128), 256, 0, stream>>>(gX2, dsl, lr, g1b, v1b, ZL128);

  // Base1 = Q_head @ W1^T ; layer-1 decayed attention -> Z1_
  gemm(stream, Qbf, W1bf, Bas1, nullptr, 1024, 128, 64, 1024, 64, 128,
       1048576, 64, 131072, 8192, 2097152, 131072, 0, 1, 32);
  k_attn<64, 128><<<dim3(16, 32), 128, 0, stream>>>(Qbf, 1024, 1048576, 64,
                                                    Kbf, 1024, 1048576, 64,
                                                    v1b, Bas1, cum, Z1_);
  k_silu<<<ew(ZL128), 256, 0, stream>>>(Z1_, X2_b, ZL128);

  // Base2 = X2_ @ W2^T ; layer-2 decayed attention -> Z2_
  gemm(stream, X2_b, W2bf, Bas2, nullptr, 1024, 64, 128, 128, 128, 64,
       2097152, 131072, 131072, 8192, 1048576, 65536, 0, 1, 32);
  k_attn<128, 64><<<dim3(16, 32), 128, 0, stream>>>(X2_b, 128, 2097152, 131072,
                                                    X2b, 128, 2097152, 131072,
                                                    v2b, Bas2, cum, Z2_);

  // W1_next / W2_next
  k_u<<<ew(ZL128), 256, 0, stream>>>(g1b, lr, cum, u1b, ZL128, 7);
  k_u<<<ew(ZL64), 256, 0, stream>>>(gz2b, lr, cum, u2b, ZL64, 6);
  gemm(stream, u1b, Kbf, W1r, nullptr, 128, 64, 1024, 128, 1024, 64,
       2097152, 131072, 1048576, 64, 131072, 8192, 1, 0, 32);
  gemm(stream, u2b, X2b, W2r, nullptr, 64, 128, 1024, 64, 128, 128,
       1048576, 65536, 2097152, 131072, 131072, 8192, 1, 0, 32);
  k_wnext<<<ew(262144), 256, 0, stream>>>(W1r, W1, cum, W1n, 262144);
  k_wnext<<<ew(262144), 256, 0, stream>>>(W2r, W2, cum, W2n, 262144);

  // output projection
  k_z2cat<<<ew(ZL64), 256, 0, stream>>>(Z2_, z2c, ZL64);
  gemm(stream, z2c, owbf, out, o_b, 2048, 1024, 1024, 1024, 1024, 1024,
       0,0,0,0,0,0, 0, 1, 1);
}
